// MultiHeadNeighborhoodAggregator_6923487282092
// MI455X (gfx1250) — matrix operand order repaired and checked by reading, not compiled, not run on hardware
//
#include <hip/hip_runtime.h>
#include <hip/hip_bf16.h>

// ---------------------------------------------------------------------------
// MultiHeadNeighborhoodAggregator, fused, bf16 WMMA path for gfx1250 (CDNA5)
// B=32768 rows, K=64 neighbors, D=256, H=2 heads, DA=32, DH=128
// 4 rows per block, 8 wave32s. All GEMMs (K/Q/V/gate) on v_wmma_f32_16x16x32_bf16.
// ---------------------------------------------------------------------------

#define BB 32768
#define KK 64
#define DD 256
#define ROWS 4

typedef __attribute__((ext_vector_type(16))) __bf16          v16bf;
typedef __attribute__((ext_vector_type(8)))  float           v8f;
typedef __attribute__((ext_vector_type(16))) unsigned short  v16u;
typedef __attribute__((ext_vector_type(8)))  unsigned short  v8u;

__device__ __forceinline__ unsigned f2bf(float f) {
    unsigned u = __float_as_uint(f);
    unsigned r = u + 0x7FFFu + ((u >> 16) & 1u);   // round-to-nearest-even
    return r >> 16;
}
__device__ __forceinline__ float bf2f(unsigned short h) {
    return __uint_as_float(((unsigned)h) << 16);
}

// A-operand gather, contiguous K order (matches B loads):
// lanes 0-15 row m, K {d0..d0+15}; lanes 16-31 K {d0+16..d0+31}.
__device__ __forceinline__ v16u loadA(const unsigned short* rp) {
    v8u c0 = *reinterpret_cast<const v8u*>(rp);
    v8u c1 = *reinterpret_cast<const v8u*>(rp + 8);
    v16u a;
#pragma unroll
    for (int i = 0; i < 8; ++i) { a[i] = c0[i]; a[i + 8] = c1[i]; }
    return a;
}

__device__ __forceinline__ v8f wmma_bf16(v16u a, v16u b, v8f c) {
    return __builtin_amdgcn_wmma_f32_16x16x32_bf16(
        false, __builtin_bit_cast(v16bf, a),
        false, __builtin_bit_cast(v16bf, b),
        (short)0, c, false, false);
}

// --- prep: f32 -> bf16 weight conversion into workspace ---------------------
__global__ void cvt_f32_bf16(const float* __restrict__ src,
                             unsigned short* __restrict__ dst, int n) {
    int i = blockIdx.x * blockDim.x + threadIdx.x;
    if (i < n) dst[i] = (unsigned short)f2bf(src[i]);
}

// --- main fused kernel: 4 rows per block, 8 wave32s -------------------------
__global__ __launch_bounds__(256)
void mhna_kernel(const float* __restrict__ center,
                 const float* __restrict__ neigh,
                 const float* __restrict__ wts,
                 const float* __restrict__ bgate,
                 const unsigned short* __restrict__ wq_bf,
                 const unsigned short* __restrict__ wk_bf,
                 const unsigned short* __restrict__ wv_bf,
                 const unsigned short* __restrict__ wg_bf,
                 float* __restrict__ out) {
    __shared__ unsigned short nb[ROWS * KK][DD + 8];  // 256 neighbor rows, bf16
    __shared__ unsigned short gateA[16][520];         // [center|ctx] bf16, M padded to 16
    __shared__ float wcoef[ROWS * KK];
    __shared__ float qs[ROWS][KK];                    // q projections
    __shared__ float attnL[ROWS][2][KK];
    __shared__ float ctx[ROWS][DD];

    const int t    = threadIdx.x;
    const int lane = t & 31;
    const int wave = t >> 5;                          // 0..7
    const int lrow = lane & 15;
    const int lhi  = (lane >> 4) & 1;
    const int b    = blockIdx.x;                      // covers rows b*4 .. b*4+3

    // ---- phase 0: stage neighbors bf16, weights coef, center into gateA ----
    {
        const float4* ne4 = reinterpret_cast<const float4*>(
            neigh + (size_t)b * ROWS * KK * DD);
        for (int i = t; i < (ROWS * KK * DD) / 4; i += 256) {
            float4 v = ne4[i];
            int g = i >> 6;                 // global neighbor row 0..255
            int col = (i & 63) * 4;
            uint2 p;
            p.x = f2bf(v.x) | (f2bf(v.y) << 16);
            p.y = f2bf(v.z) | (f2bf(v.w) << 16);
            *reinterpret_cast<uint2*>(&nb[g][col]) = p;
        }
        wcoef[t] = wts[(size_t)b * (ROWS * KK) + t];
        // zero pad rows 4..15 of gateA (full 520-col rows, as dwords)
        for (int i = t; i < 12 * 260; i += 256) {
            int rr = 4 + i / 260, cc = (i % 260) * 2;
            *reinterpret_cast<unsigned*>(&gateA[rr][cc]) = 0u;
        }
        // center bf16 into gateA rows 0..3, cols 0..255
        for (int e = t; e < ROWS * DD; e += 256) {
            int rr = e >> 8, cc = e & 255;
            gateA[rr][cc] =
                (unsigned short)f2bf(center[((size_t)b * ROWS + rr) * DD + cc]);
        }
    }
    __syncthreads();

    // ---- phase 1: q projection via WMMA (M padded 4->16), waves 0..3 -------
    if (wave < 4) {
        v8f qacc = {};
#pragma unroll
        for (int ks = 0; ks < 8; ++ks) {
            v16u a = loadA(&gateA[lrow][ks * 32 + lhi * 16]);
            v16u bm = *reinterpret_cast<const v16u*>(
                wq_bf + (wave * 16 + lrow) * DD + ks * 32 + lhi * 16);
            qacc = wmma_bf16(a, bm, qacc);
        }
        if (lhi == 0) {
#pragma unroll
            for (int r = 0; r < ROWS; ++r) qs[r][wave * 16 + lrow] = qacc[r];
        }
    }
    __syncthreads();

    // ---- phase 2: K projection (2 m-tiles per wave) + in-register logits ---
#pragma unroll
    for (int mi = 0; mi < 2; ++mi) {
        const int mt  = 2 * wave + mi;       // m-tile 0..15 (16 neighbors each)
        const int row = mt >> 2;             // batch row of this tile
        v16u aT[8];
#pragma unroll
        for (int ks = 0; ks < 8; ++ks)
            aT[ks] = loadA(&nb[mt * 16 + lrow][ks * 32 + lhi * 16]);
        v8f kacc[4] = {};
#pragma unroll
        for (int ks = 0; ks < 8; ++ks) {
#pragma unroll
            for (int nt = 0; nt < 4; ++nt) {
                v16u bm = *reinterpret_cast<const v16u*>(
                    wk_bf + (nt * 16 + lrow) * DD + ks * 32 + lhi * 16);
                kacc[nt] = wmma_bf16(aT[ks], bm, kacc[nt]);
            }
        }
        // logits: dot(q_h, k_h) folded straight out of accumulators
#pragma unroll
        for (int h = 0; h < 2; ++h) {
            float qv0 = qs[row][h * 32 + lrow];
            float qv1 = qs[row][h * 32 + 16 + lrow];
#pragma unroll
            for (int r = 0; r < 8; ++r) {
                float p = kacc[2 * h][r] * qv0 + kacc[2 * h + 1][r] * qv1;
                p += __shfl_xor(p, 1);
                p += __shfl_xor(p, 2);
                p += __shfl_xor(p, 4);
                p += __shfl_xor(p, 8);
                if (lrow == 0) {
                    int k = (mt & 3) * 16 + r + 8 * lhi;
                    attnL[row][h][k] =
                        p * 0.17677669529663687f * wcoef[row * KK + k];
                }
            }
        }
    }
    __syncthreads();

    // ---- phase 3: softmax per (row, head) ----------------------------------
    if (t < ROWS * 2) {
        const int row = t >> 1, h = t & 1;
        float mx = -1e30f;
        for (int k = 0; k < KK; ++k) mx = fmaxf(mx, attnL[row][h][k]);
        float sum = 0.f;
        for (int k = 0; k < KK; ++k) {
            float e = __expf(attnL[row][h][k] - mx);
            attnL[row][h][k] = e;
            sum += e;
        }
        float inv = 1.f / sum;
        for (int k = 0; k < KK; ++k) attnL[row][h][k] *= inv;
    }
    __syncthreads();

    // ---- phase 4: V projection; wave owns 2 output n-tiles for ALL rows ----
    // B tiles cached in registers once -> V weights read once per block.
    {
        v16u Bc[2][8];
#pragma unroll
        for (int ntl = 0; ntl < 2; ++ntl) {
            const int nt = 2 * wave + ntl;
#pragma unroll
            for (int ks = 0; ks < 8; ++ks)
                Bc[ntl][ks] = *reinterpret_cast<const v16u*>(
                    wv_bf + (nt * 16 + lrow) * DD + ks * 32 + lhi * 16);
        }
        for (int row = 0; row < ROWS; ++row) {
            float cacc0 = 0.f, cacc1 = 0.f;
#pragma unroll
            for (int mtl = 0; mtl < 4; ++mtl) {
                const int mt = row * 4 + mtl;
                v16u aT[8];
#pragma unroll
                for (int ks = 0; ks < 8; ++ks)
                    aT[ks] = loadA(&nb[mt * 16 + lrow][ks * 32 + lhi * 16]);
#pragma unroll
                for (int ntl = 0; ntl < 2; ++ntl) {
                    v8f vacc = {};
#pragma unroll
                    for (int ks = 0; ks < 8; ++ks)
                        vacc = wmma_bf16(aT[ks], Bc[ntl][ks], vacc);
                    const int h = (2 * wave + ntl) >> 3;   // head of this col
                    float s = 0.f;
#pragma unroll
                    for (int r = 0; r < 8; ++r)
                        s += attnL[row][h][mtl * 16 + r + 8 * lhi] * vacc[r];
                    if (ntl == 0) cacc0 += s; else cacc1 += s;
                }
            }
            // merge lane halves (same column, disjoint neighbor subsets)
#pragma unroll
            for (int ntl = 0; ntl < 2; ++ntl) {
                float cv = (ntl == 0) ? cacc0 : cacc1;
                float v = cv + __shfl_xor(cv, 16);
                if (lhi == 0) {
                    const int col = (2 * wave + ntl) * 16 + lrow;
                    ctx[row][col] = v;
                    gateA[row][DD + col] = (unsigned short)f2bf(v);
                }
            }
        }
    }
    __syncthreads();

    // ---- phase 5: gate GEMM via WMMA (M padded 4->16, K=512) + epilogue ----
    {
        v8f gacc[2] = {};
#pragma unroll
        for (int ks = 0; ks < 16; ++ks) {
            v16u a = loadA(&gateA[lrow][ks * 32 + lhi * 16]);
#pragma unroll
            for (int ntl = 0; ntl < 2; ++ntl) {
                v16u bm = *reinterpret_cast<const v16u*>(
                    wg_bf + ((2 * wave + ntl) * 16 + lrow) * (2 * DD) +
                    ks * 32 + lhi * 16);
                gacc[ntl] = wmma_bf16(a, bm, gacc[ntl]);
            }
        }
        if (lhi == 0) {
#pragma unroll
            for (int ntl = 0; ntl < 2; ++ntl) {
                const int col = (2 * wave + ntl) * 16 + lrow;
                const float bgv = bgate[col];
#pragma unroll
                for (int r = 0; r < ROWS; ++r) {
                    float s = gacc[ntl][r] + bgv;
                    float g = 1.f / (1.f + __expf(-s));
                    const size_t o = ((size_t)b * ROWS + r) * DD + col;
                    float cen = center[o];
                    out[o] = g * cen + (1.f - g) * ctx[r][col];
                }
            }
        }
    }
}

// ---------------------------------------------------------------------------
extern "C" void kernel_launch(void* const* d_in, const int* in_sizes, int n_in,
                              void* d_out, int out_size, void* d_ws, size_t ws_size,
                              hipStream_t stream) {
    const float* center = (const float*)d_in[0];
    const float* neigh  = (const float*)d_in[1];
    const float* wts    = (const float*)d_in[2];
    const float* W_q    = (const float*)d_in[3];
    const float* W_k    = (const float*)d_in[4];
    const float* W_v    = (const float*)d_in[5];
    const float* W_gate = (const float*)d_in[6];
    const float* b_gate = (const float*)d_in[7];
    float* out = (float*)d_out;

    unsigned short* wsb = (unsigned short*)d_ws;
    unsigned short* wq_bf = wsb;                         // 64*256
    unsigned short* wk_bf = wq_bf + 64 * 256;            // 64*256
    unsigned short* wv_bf = wk_bf + 64 * 256;            // 256*256
    unsigned short* wg_bf = wv_bf + 256 * 256;           // 256*512

    cvt_f32_bf16<<<(64 * 256 + 255) / 256, 256, 0, stream>>>(W_q, wq_bf, 64 * 256);
    cvt_f32_bf16<<<(64 * 256 + 255) / 256, 256, 0, stream>>>(W_k, wk_bf, 64 * 256);
    cvt_f32_bf16<<<(256 * 256 + 255) / 256, 256, 0, stream>>>(W_v, wv_bf, 256 * 256);
    cvt_f32_bf16<<<(256 * 512 + 255) / 256, 256, 0, stream>>>(W_gate, wg_bf, 256 * 512);

    mhna_kernel<<<BB / ROWS, 256, 0, stream>>>(center, neigh, wts, b_gate,
                                               wq_bf, wk_bf, wv_bf, wg_bf, out);
}
